// MambaBlock_15934328668934
// MI455X (gfx1250) — matrix operand order repaired and checked by reading, not compiled, not run on hardware
//
#include <hip/hip_runtime.h>
#include <hip/hip_bf16.h>
#include <math.h>

// ---------------------------------------------------------------------------
// Mamba block (3 layers) for gfx1250 (MI455X): f16 WMMA GEMMs + fp32 scan.
// Wave tile = 16x64 (4 WMMA subtiles sharing one A fragment). All N dims are
// padded to multiples of 64 so the K-loop has NO branches around the WMMAs.
// ---------------------------------------------------------------------------

typedef __attribute__((ext_vector_type(16))) _Float16 v16h;
typedef __attribute__((ext_vector_type(8)))  float    v8f;

union HFrag { v16h v; _Float16 h[16]; };
union CFrag { v8f  v; float     f[8];  };

// Dims
#define BATCH   16
#define NSEQ    1024
#define IN_DIM  192
#define D_MODEL 256
#define DI      512
#define ML      (BATCH * NSEQ)   // 16384 rows
#define XDBL_LD 64               // padded x_dbl row: dt[0:16], B[16:32], C[32:48], 0[48:64]

// ---------------------------------------------------------------------------
// GEMM: Y[M,N] = X[M,K] (row stride ldx) * W[N,K]^T, f16 WMMA, f32 acc.
// ep: 0 = none, 1 = +bias, 2 = +bias then softplus
// Block: 256 threads = 8 waves, arranged 4 (M) x 2 (N-groups of 64).
// Requires: M % 64 == 0, N % 64 == 0. K arbitrary (masked tail).
// ---------------------------------------------------------------------------
__global__ __launch_bounds__(256) void gemm_wmma_kernel(
    const float* __restrict__ X, int ldx,
    const float* __restrict__ W,
    const float* __restrict__ bias,
    float* __restrict__ Y,
    int M, int N, int K, int ep)
{
    const int tid  = threadIdx.x;
    const int lane = tid & 31;
    // wave id as an SGPR -> whole-wave scalar guard
    const int wave = __builtin_amdgcn_readfirstlane(tid) >> 5;

    const int m0 = blockIdx.x * 64 + (wave & 3) * 16;
    const int n0 = blockIdx.y * 128 + (wave >> 2) * 64;
    if (m0 >= M || n0 >= N) return;          // scalar, whole-wave

    const int  lhalf = lane & 15;
    const bool hi    = lane >= 16;

    // A fragment: row = m0+lhalf; halves 0..7 -> K = k0+akoff+j, 8..15 -> +16
    const int akoff = hi ? 8 : 0;
    // B fragment: col; halves 0..7 -> K = k0+bkoff+j, 8..15 -> +16
    const int bkoff = hi ? 8 : 0;

    const float* __restrict__ Xr  = X + (size_t)(m0 + lhalf) * ldx;
    const float* __restrict__ Wr0 = W + (size_t)(n0 + lhalf) * K;
    const size_t wstep = (size_t)16 * K;     // 16 columns of W per sub-tile

    CFrag acc[4];
#pragma unroll
    for (int i = 0; i < 4; ++i)
#pragma unroll
        for (int r = 0; r < 8; ++r) acc[i].f[r] = 0.0f;

    for (int k0 = 0; k0 < K; k0 += 32) {
        HFrag a, bfr[4];
        if (k0 + 32 <= K) {
            // ---- fast path: b128 loads (strides here are 16B-aligned) ----
            const float4 a0 = *(const float4*)(Xr + k0 + akoff);
            const float4 a1 = *(const float4*)(Xr + k0 + akoff + 4);
            const float4 a2 = *(const float4*)(Xr + k0 + 16 + akoff);
            const float4 a3 = *(const float4*)(Xr + k0 + 16 + akoff + 4);
            a.h[0]=(_Float16)a0.x;  a.h[1]=(_Float16)a0.y;  a.h[2]=(_Float16)a0.z;  a.h[3]=(_Float16)a0.w;
            a.h[4]=(_Float16)a1.x;  a.h[5]=(_Float16)a1.y;  a.h[6]=(_Float16)a1.z;  a.h[7]=(_Float16)a1.w;
            a.h[8]=(_Float16)a2.x;  a.h[9]=(_Float16)a2.y;  a.h[10]=(_Float16)a2.z; a.h[11]=(_Float16)a2.w;
            a.h[12]=(_Float16)a3.x; a.h[13]=(_Float16)a3.y; a.h[14]=(_Float16)a3.z; a.h[15]=(_Float16)a3.w;

#pragma unroll
            for (int i = 0; i < 4; ++i) {
                const float* __restrict__ Wr = Wr0 + (size_t)i * wstep;
                const float4 b0 = *(const float4*)(Wr + k0 + bkoff);
                const float4 b1 = *(const float4*)(Wr + k0 + bkoff + 4);
                const float4 b2 = *(const float4*)(Wr + k0 + bkoff + 16);
                const float4 b3 = *(const float4*)(Wr + k0 + bkoff + 20);
                bfr[i].h[0]=(_Float16)b0.x;  bfr[i].h[1]=(_Float16)b0.y;
                bfr[i].h[2]=(_Float16)b0.z;  bfr[i].h[3]=(_Float16)b0.w;
                bfr[i].h[4]=(_Float16)b1.x;  bfr[i].h[5]=(_Float16)b1.y;
                bfr[i].h[6]=(_Float16)b1.z;  bfr[i].h[7]=(_Float16)b1.w;
                bfr[i].h[8]=(_Float16)b2.x;  bfr[i].h[9]=(_Float16)b2.y;
                bfr[i].h[10]=(_Float16)b2.z; bfr[i].h[11]=(_Float16)b2.w;
                bfr[i].h[12]=(_Float16)b3.x; bfr[i].h[13]=(_Float16)b3.y;
                bfr[i].h[14]=(_Float16)b3.z; bfr[i].h[15]=(_Float16)b3.w;
            }
            if (k0 + 32 < K) {               // speculative prefetch of next K chunk
                __builtin_prefetch(Xr + k0 + 32, 0, 1);
                __builtin_prefetch(Wr0 + k0 + 32, 0, 1);
            }
        } else {
            // ---- masked tail (dt_proj: K=16) ----
#pragma unroll
            for (int j = 0; j < 8; ++j) {
                const int ka = k0 + akoff + j;
                const int kb = k0 + 16 + akoff + j;
                a.h[j]     = (_Float16)((ka < K) ? Xr[ka] : 0.0f);
                a.h[j + 8] = (_Float16)((kb < K) ? Xr[kb] : 0.0f);
            }
#pragma unroll
            for (int i = 0; i < 4; ++i) {
                const float* __restrict__ Wr = Wr0 + (size_t)i * wstep;
#pragma unroll
                for (int j = 0; j < 16; ++j) {
                    const int kb = k0 + bkoff + (j < 8 ? j : j + 8);
                    bfr[i].h[j] = (_Float16)((kb < K) ? Wr[kb] : 0.0f);
                }
            }
        }

        // 4 WMMAs sharing the A fragment, no branches in between
#pragma unroll
        for (int i = 0; i < 4; ++i) {
            acc[i].v = __builtin_amdgcn_wmma_f32_16x16x32_f16(
                /*neg_a=*/false, a.v, /*neg_b=*/false, bfr[i].v,
                /*c_mod=*/(short)0, acc[i].v,
                /*reuse_a=*/false, /*reuse_b=*/false);
        }
    }

    // C/D layout: lane(0-15): N=lhalf, M=m0+r ; lane(16-31): M=m0+8+r
    const int mbase = m0 + (hi ? 8 : 0);
#pragma unroll
    for (int i = 0; i < 4; ++i) {
        const int ncol = n0 + i * 16 + lhalf;
        const float bval = (ep >= 1 && bias) ? bias[ncol] : 0.0f;
#pragma unroll
        for (int r = 0; r < 8; ++r) {
            float v = acc[i].f[r] + bval;
            if (ep == 2) v = (v > 20.0f) ? v : logf(1.0f + __expf(v));  // softplus
            Y[(size_t)(mbase + r) * N + ncol] = v;
        }
    }
}

// ---------------------------------------------------------------------------
// Zero-pad x_proj weights: (48,512) -> (64,512) with rows 48..63 = 0.
// ---------------------------------------------------------------------------
__global__ __launch_bounds__(256) void pad_w_kernel(
    const float* __restrict__ src, float* __restrict__ dst)
{
    const int idx = blockIdx.x * 256 + threadIdx.x;   // 64*512 elements
    const int row = idx >> 9;
    const int col = idx & (DI - 1);
    dst[idx] = (row < 48) ? src[row * DI + col] : 0.0f;
}

// ---------------------------------------------------------------------------
// Depthwise causal conv (width 4) + SiLU.  xz layout (B,L,2*DI); xc out (B,L,DI)
// ---------------------------------------------------------------------------
__global__ __launch_bounds__(256) void conv_silu_kernel(
    const float* __restrict__ xz, const float* __restrict__ cw,
    const float* __restrict__ cb, float* __restrict__ xc, int total)
{
    const int idx = blockIdx.x * 256 + threadIdx.x;
    if (idx >= total) return;
    const int d   = idx & (DI - 1);
    const int t   = (idx >> 9) & (NSEQ - 1);
    const int row = idx >> 9;                  // b*L + t
    float acc = cb[d];
#pragma unroll
    for (int j = 0; j < 4; ++j) {
        const int tt = t - 3 + j;
        if (tt >= 0)
            acc = fmaf(cw[d * 4 + j], xz[(size_t)(row - 3 + j) * (2 * DI) + d], acc);
    }
    xc[idx] = acc / (1.0f + __expf(-acc));     // silu
}

// ---------------------------------------------------------------------------
// Selective scan: grid (BATCH, DI/128), 128 threads = one channel each.
// h[16] state in registers; B/C broadcast-loaded per timestep.
// ---------------------------------------------------------------------------
__global__ __launch_bounds__(128) void scan_kernel(
    const float* __restrict__ xc,   const float* __restrict__ delta,
    const float* __restrict__ xdbl, const float* __restrict__ A_log,
    const float* __restrict__ Dp,   float* __restrict__ y)
{
    const int b = blockIdx.x;
    const int d = blockIdx.y * 128 + threadIdx.x;
    float A[16], h[16];
#pragma unroll
    for (int s = 0; s < 16; ++s) { A[s] = -__expf(A_log[d * 16 + s]); h[s] = 0.0f; }
    const float Dv = Dp[d];

    for (int t = 0; t < NSEQ; ++t) {
        const size_t row = (size_t)b * NSEQ + t;
        const float dt = delta[row * DI + d];
        const float u  = xc[row * DI + d];
        const float* __restrict__ bc = xdbl + row * XDBL_LD;
        const float du = dt * u;
        float yv = u * Dv;
#pragma unroll
        for (int s = 0; s < 16; ++s) {
            const float dA = __expf(dt * A[s]);
            h[s] = fmaf(dA, h[s], du * bc[16 + s]);
            yv   = fmaf(h[s], bc[32 + s], yv);
        }
        y[row * DI + d] = yv;
    }
}

// ---------------------------------------------------------------------------
// y *= silu(z),  z = xz[..., DI:2*DI]
// ---------------------------------------------------------------------------
__global__ __launch_bounds__(256) void gate_kernel(
    float* __restrict__ y, const float* __restrict__ xz, int total)
{
    const int idx = blockIdx.x * 256 + threadIdx.x;
    if (idx >= total) return;
    const int d   = idx & (DI - 1);
    const int row = idx >> 9;
    const float z = xz[(size_t)row * (2 * DI) + DI + d];
    y[idx] *= z / (1.0f + __expf(-z));
}

// ---------------------------------------------------------------------------

static inline void launch_gemm(const float* X, int ldx, const float* W,
                               const float* bias, float* Y,
                               int M, int N, int K, int ep, hipStream_t s)
{
    dim3 grid(M / 64, (N + 127) / 128);
    gemm_wmma_kernel<<<grid, 256, 0, s>>>(X, ldx, W, bias, Y, M, N, K, ep);
}

extern "C" void kernel_launch(void* const* d_in, const int* in_sizes, int n_in,
                              void* d_out, int out_size, void* d_ws, size_t ws_size,
                              hipStream_t stream)
{
    const float* x    = (const float*)d_in[0];   // (B,L,192)
    const float* ipw  = (const float*)d_in[1];   // (256,192)
    const float* ipb  = (const float*)d_in[2];   // (256,)
    const float* inw  = (const float*)d_in[3];   // (3,1024,256)
    const float* cw   = (const float*)d_in[4];   // (3,512,4)
    const float* cb   = (const float*)d_in[5];   // (3,512)
    const float* xpw  = (const float*)d_in[6];   // (3,48,512)
    const float* dtw  = (const float*)d_in[7];   // (3,512,16)
    const float* dtb  = (const float*)d_in[8];   // (3,512)
    const float* alog = (const float*)d_in[9];   // (3,512,16)
    const float* Dpar = (const float*)d_in[10];  // (3,512)
    const float* outw = (const float*)d_in[11];  // (3,256,512)

    float* ws     = (float*)d_ws;
    float* h      = ws;  ws += (size_t)ML * D_MODEL;    // 16 MB
    float* xz     = ws;  ws += (size_t)ML * 2 * DI;     // 64 MB
    float* xc     = ws;  ws += (size_t)ML * DI;         // 32 MB
    float* xdbl   = ws;  ws += (size_t)ML * XDBL_LD;    //  4 MB
    float* delta  = ws;  ws += (size_t)ML * DI;         // 32 MB
    float* ybuf   = ws;  ws += (size_t)ML * DI;         // 32 MB
    float* xpwpad = ws;                                 // 64*512 floats

    const int ew_blocks = (ML * DI) / 256;

    // input projection: h = x @ ipw^T + ipb
    launch_gemm(x, IN_DIM, ipw, ipb, h, ML, D_MODEL, IN_DIM, 1, stream);

    for (int l = 0; l < 3; ++l) {
        const float* inw_l  = inw  + (size_t)l * 2 * DI * D_MODEL;
        const float* cw_l   = cw   + (size_t)l * DI * 4;
        const float* cb_l   = cb   + (size_t)l * DI;
        const float* xpw_l  = xpw  + (size_t)l * 48 * DI;
        const float* dtw_l  = dtw  + (size_t)l * DI * 16;
        const float* dtb_l  = dtb  + (size_t)l * DI;
        const float* alog_l = alog + (size_t)l * DI * 16;
        const float* Dp_l   = Dpar + (size_t)l * DI;
        const float* outw_l = outw + (size_t)l * D_MODEL * DI;

        // xz = h @ in_w^T                        (16384 x 256 -> 1024)
        launch_gemm(h, D_MODEL, inw_l, nullptr, xz, ML, 2 * DI, D_MODEL, 0, stream);
        // xc = silu(causal_conv(xz[:, :DI]))
        conv_silu_kernel<<<ew_blocks, 256, 0, stream>>>(xz, cw_l, cb_l, xc, ML * DI);
        // pad x_proj weights to 64 rows, then x_dbl = xc @ xp_w^T (N=64 padded)
        pad_w_kernel<<<(XDBL_LD * DI) / 256, 256, 0, stream>>>(xpw_l, xpwpad);
        launch_gemm(xc, DI, xpwpad, nullptr, xdbl, ML, XDBL_LD, DI, 0, stream);
        // delta = softplus(dt @ dtp_w^T + dtp_b) (16384 x 16 -> 512), dt = xdbl[:, :16]
        launch_gemm(xdbl, XDBL_LD, dtw_l, dtb_l, delta, ML, DI, 16, 2, stream);
        // selective scan
        scan_kernel<<<dim3(BATCH, DI / 128), 128, 0, stream>>>(
            xc, delta, xdbl, alog_l, Dp_l, ybuf);
        // y *= silu(z)
        gate_kernel<<<ew_blocks, 256, 0, stream>>>(ybuf, xz, ML * DI);
        // h = y @ out_w^T                        (16384 x 512 -> 256)
        float* hout = (l == 2) ? (float*)d_out : h;
        launch_gemm(ybuf, DI, outw_l, nullptr, hout, ML, D_MODEL, DI, 0, stream);
    }

    (void)in_sizes; (void)n_in; (void)out_size; (void)ws_size;
}
